// GRUCell_87007447482533
// MI455X (gfx1250) — matrix operand order repaired and checked by reading, not compiled, not run on hardware
//
#include <hip/hip_runtime.h>
#include <hip/hip_bf16.h>
#include <cstdint>
#include <cstddef>

#define B_DIM  4096
#define IN_DIM 1024
#define H_DIM  4096
#define EPSV   1e-5f

typedef unsigned short u16;
typedef __attribute__((ext_vector_type(16))) __bf16 v16bf;
typedef __attribute__((ext_vector_type(8)))  float  v8f;

union FragU { v16bf v; uint4 q[2]; };

// A fragment: 16x32 bf16, rows m..m+15, K chunk k0..k0+31 of row-major matrix (ld elems)
// lane l (l16=l&15, half=l>>4) holds row m+l16:
//   VGPRs 0-3: K = k0 + half*8 + 0..7 ; VGPRs 4-7: K = k0 + 16 + half*8 + 0..7
__device__ __forceinline__ v16bf load_a_frag(const u16* __restrict__ m, int ld, int row,
                                             int k0, int half) {
    FragU f;
    const u16* p = m + (size_t)row * ld + k0 + half * 8;
    f.q[0] = *(const uint4*)(p);
    f.q[1] = *(const uint4*)(p + 16);
    return f.v;
}

// B fragment: 32x16 bf16 (K x N) where B = W^T, W row-major (ld elems).
// lane l holds column n = n0+l16 (= W row n), K = k0 + half*8 + 0..7, then k0 + 16 + half*8 + 0..7
__device__ __forceinline__ v16bf load_b_frag(const u16* __restrict__ w, int ld, int col,
                                             int k0, int half) {
    FragU f;
    const u16* p = w + (size_t)col * ld + k0 + half * 8;
    f.q[0] = *(const uint4*)(p);
    f.q[1] = *(const uint4*)(p + 16);
    return f.v;
}

__device__ __forceinline__ v8f wmma_bf16(v16bf a, v16bf b, v8f c) {
    return __builtin_amdgcn_wmma_f32_16x16x32_bf16(false, a, false, b, (short)0, c,
                                                   false, false);
}

__device__ __forceinline__ float sigmoidf_fast(float x) {
    return 1.0f / (1.0f + __expf(-x));
}

// ---------------------------------------------------------------------------
// f32 -> bf16 (round-to-nearest-even) cast
// ---------------------------------------------------------------------------
__global__ void cast_f32_to_bf16(const float* __restrict__ src, u16* __restrict__ dst,
                                 size_t n) {
    size_t i = (size_t)blockIdx.x * blockDim.x + threadIdx.x;
    size_t stride = (size_t)gridDim.x * blockDim.x;
    for (; i < n; i += stride) {
        uint32_t u = __float_as_uint(src[i]);
        uint32_t r = u + 0x7FFFu + ((u >> 16) & 1u);
        dst[i] = (u16)(r >> 16);
    }
}

// ---------------------------------------------------------------------------
// Fused 6-GEMM gate kernel.
// Block: 256 threads = 8 waves, block tile 128(M) x 64(N); wave tile 32x32
// (2x2 grid of 16x16 WMMA tiles). Four f32 accumulator sets per tile:
//   acc_r = x@Wr^T + h@Ur^T     acc_z = x@Wz^T + h@Uz^T
//   acc_w = x@Wh^T              acc_u = h@Uh^T
// Epilogue: r=sig(acc_r+br), z=sig(acc_z+bz), h_pre=acc_w+bw + r*(acc_u+bu)
// ---------------------------------------------------------------------------
__global__ __launch_bounds__(256) void gru_gates_wmma(
    const u16* __restrict__ xbf,                     // B x IN
    const u16* __restrict__ hbf,                     // B x H
    const u16* __restrict__ wr, const u16* __restrict__ wz, const u16* __restrict__ wh,
    const u16* __restrict__ ur, const u16* __restrict__ uz, const u16* __restrict__ uh,
    const float* __restrict__ wrb, const float* __restrict__ urb,
    const float* __restrict__ wzb, const float* __restrict__ uzb,
    const float* __restrict__ whb, const float* __restrict__ uhb,
    float* __restrict__ zbuf, float* __restrict__ hprebuf) {
    const int lane = threadIdx.x & 31;
    const int wid  = threadIdx.x >> 5;
    const int half = lane >> 4;
    const int l16  = lane & 15;
    const int wm   = wid & 3;          // 4 waves along M
    const int wn   = wid >> 2;         // 2 waves along N
    const int m_base = blockIdx.y * 128 + wm * 32;
    const int n_base = blockIdx.x * 64  + wn * 32;

    v8f acc_r[2][2] = {}; v8f acc_z[2][2] = {};
    v8f acc_w[2][2] = {}; v8f acc_u[2][2] = {};

    const int row0 = m_base + l16;
    const int row1 = m_base + 16 + l16;
    const int col0 = n_base + l16;
    const int col1 = n_base + 16 + l16;

    // ---- Loop 1: K over IN (x side), gates r, z, w(h) -----------------
    for (int k0 = 0; k0 < IN_DIM; k0 += 32) {
        v16bf a0 = load_a_frag(xbf, IN_DIM, row0, k0, half);
        v16bf a1 = load_a_frag(xbf, IN_DIM, row1, k0, half);
        {
            v16bf b0 = load_b_frag(wr, IN_DIM, col0, k0, half);
            v16bf b1 = load_b_frag(wr, IN_DIM, col1, k0, half);
            acc_r[0][0] = wmma_bf16(a0, b0, acc_r[0][0]);
            acc_r[0][1] = wmma_bf16(a0, b1, acc_r[0][1]);
            acc_r[1][0] = wmma_bf16(a1, b0, acc_r[1][0]);
            acc_r[1][1] = wmma_bf16(a1, b1, acc_r[1][1]);
        }
        {
            v16bf b0 = load_b_frag(wz, IN_DIM, col0, k0, half);
            v16bf b1 = load_b_frag(wz, IN_DIM, col1, k0, half);
            acc_z[0][0] = wmma_bf16(a0, b0, acc_z[0][0]);
            acc_z[0][1] = wmma_bf16(a0, b1, acc_z[0][1]);
            acc_z[1][0] = wmma_bf16(a1, b0, acc_z[1][0]);
            acc_z[1][1] = wmma_bf16(a1, b1, acc_z[1][1]);
        }
        {
            v16bf b0 = load_b_frag(wh, IN_DIM, col0, k0, half);
            v16bf b1 = load_b_frag(wh, IN_DIM, col1, k0, half);
            acc_w[0][0] = wmma_bf16(a0, b0, acc_w[0][0]);
            acc_w[0][1] = wmma_bf16(a0, b1, acc_w[0][1]);
            acc_w[1][0] = wmma_bf16(a1, b0, acc_w[1][0]);
            acc_w[1][1] = wmma_bf16(a1, b1, acc_w[1][1]);
        }
    }

    // ---- Loop 2: K over H (hidden side), gates r, z, u(h) -------------
    for (int k0 = 0; k0 < H_DIM; k0 += 32) {
        v16bf a0 = load_a_frag(hbf, H_DIM, row0, k0, half);
        v16bf a1 = load_a_frag(hbf, H_DIM, row1, k0, half);
        {
            v16bf b0 = load_b_frag(ur, H_DIM, col0, k0, half);
            v16bf b1 = load_b_frag(ur, H_DIM, col1, k0, half);
            acc_r[0][0] = wmma_bf16(a0, b0, acc_r[0][0]);
            acc_r[0][1] = wmma_bf16(a0, b1, acc_r[0][1]);
            acc_r[1][0] = wmma_bf16(a1, b0, acc_r[1][0]);
            acc_r[1][1] = wmma_bf16(a1, b1, acc_r[1][1]);
        }
        {
            v16bf b0 = load_b_frag(uz, H_DIM, col0, k0, half);
            v16bf b1 = load_b_frag(uz, H_DIM, col1, k0, half);
            acc_z[0][0] = wmma_bf16(a0, b0, acc_z[0][0]);
            acc_z[0][1] = wmma_bf16(a0, b1, acc_z[0][1]);
            acc_z[1][0] = wmma_bf16(a1, b0, acc_z[1][0]);
            acc_z[1][1] = wmma_bf16(a1, b1, acc_z[1][1]);
        }
        {
            v16bf b0 = load_b_frag(uh, H_DIM, col0, k0, half);
            v16bf b1 = load_b_frag(uh, H_DIM, col1, k0, half);
            acc_u[0][0] = wmma_bf16(a0, b0, acc_u[0][0]);
            acc_u[0][1] = wmma_bf16(a0, b1, acc_u[0][1]);
            acc_u[1][0] = wmma_bf16(a1, b0, acc_u[1][0]);
            acc_u[1][1] = wmma_bf16(a1, b1, acc_u[1][1]);
        }
    }

    // ---- Epilogue: biases + sigmoid + reset-scaled candidate ----------
    // C/D layout: VGPR j, lane -> row = j + half*8, col = l16 (per 16x16 tile)
    #pragma unroll
    for (int tj = 0; tj < 2; ++tj) {
        const int n = n_base + tj * 16 + l16;
        const float rb = wrb[n] + urb[n];
        const float zb = wzb[n] + uzb[n];
        const float wb = whb[n];
        const float ub = uhb[n];
        #pragma unroll
        for (int ti = 0; ti < 2; ++ti) {
            #pragma unroll
            for (int j = 0; j < 8; ++j) {
                const int m = m_base + ti * 16 + half * 8 + j;
                const float rv = sigmoidf_fast(acc_r[ti][tj][j] + rb);
                const float zv = sigmoidf_fast(acc_z[ti][tj][j] + zb);
                const float hp = acc_w[ti][tj][j] + wb + rv * (acc_u[ti][tj][j] + ub);
                const size_t idx = (size_t)m * H_DIM + n;
                zbuf[idx]    = zv;
                hprebuf[idx] = hp;
            }
        }
    }
}

// ---------------------------------------------------------------------------
// Global mean / mean-of-squares reduction over h_pre (16.8M elements)
// ---------------------------------------------------------------------------
__global__ __launch_bounds__(256) void bn_stats(const float* __restrict__ hpre,
                                                float* __restrict__ stats) {
    __shared__ float s_sum[256];
    __shared__ float s_sq[256];
    const size_t n4 = (size_t)B_DIM * H_DIM / 4;
    float sum = 0.f, sq = 0.f;
    for (size_t i = (size_t)blockIdx.x * blockDim.x + threadIdx.x; i < n4;
         i += (size_t)gridDim.x * blockDim.x) {
        float4 v = ((const float4*)hpre)[i];
        sum += v.x + v.y + v.z + v.w;
        sq  += v.x * v.x + v.y * v.y + v.z * v.z + v.w * v.w;
    }
    s_sum[threadIdx.x] = sum;
    s_sq[threadIdx.x]  = sq;
    __syncthreads();
    for (int s = 128; s > 0; s >>= 1) {
        if ((int)threadIdx.x < s) {
            s_sum[threadIdx.x] += s_sum[threadIdx.x + s];
            s_sq[threadIdx.x]  += s_sq[threadIdx.x + s];
        }
        __syncthreads();
    }
    if (threadIdx.x == 0) {
        atomicAdd(&stats[0], s_sum[0]);
        atomicAdd(&stats[1], s_sq[0]);
    }
}

// ---------------------------------------------------------------------------
// Finalize: BN + relu + gate blend + 4-wide output projection (one block/row)
// ---------------------------------------------------------------------------
__global__ __launch_bounds__(256) void gru_finalize(
    const float* __restrict__ zbuf, const float* __restrict__ hprebuf,
    const float* __restrict__ hidden, const float* __restrict__ stats,
    const float* __restrict__ out_w, const float* __restrict__ out_b,
    const float* __restrict__ gamma, const float* __restrict__ beta,
    float* __restrict__ state_out, float* __restrict__ proj_out) {
    const int row = blockIdx.x;
    const int t   = threadIdx.x;
    const float inv_n = 1.0f / (float)((size_t)B_DIM * H_DIM);
    const float mean  = stats[0] * inv_n;
    const float var   = stats[1] * inv_n - mean * mean;
    const float scale = rsqrtf(var + EPSV) * gamma[0];
    const float shift = beta[0] - mean * scale;

    float p0 = 0.f, p1 = 0.f, p2 = 0.f, p3 = 0.f;
    for (int c = t; c < H_DIM; c += 256) {
        const size_t idx = (size_t)row * H_DIM + c;
        const float z  = zbuf[idx];
        const float hp = hprebuf[idx];
        float prop = hp * scale + shift;
        prop = prop > 0.f ? prop : 0.f;
        const float st = (1.0f - z) * hidden[idx] + z * prop;
        state_out[idx] = st;
        p0 += st * out_w[c];
        p1 += st * out_w[H_DIM + c];
        p2 += st * out_w[2 * H_DIM + c];
        p3 += st * out_w[3 * H_DIM + c];
    }
    __shared__ float red[4][256];
    red[0][t] = p0; red[1][t] = p1; red[2][t] = p2; red[3][t] = p3;
    __syncthreads();
    for (int s = 128; s > 0; s >>= 1) {
        if (t < s) {
            #pragma unroll
            for (int o = 0; o < 4; ++o) red[o][t] += red[o][t + s];
        }
        __syncthreads();
    }
    if (t < 4) proj_out[(size_t)t * B_DIM + row] = red[t][0] + out_b[t];
}

// ---------------------------------------------------------------------------
extern "C" void kernel_launch(void* const* d_in, const int* in_sizes, int n_in,
                              void* d_out, int out_size, void* d_ws, size_t ws_size,
                              hipStream_t stream) {
    (void)in_sizes; (void)n_in; (void)out_size; (void)ws_size;

    const float* x      = (const float*)d_in[0];
    const float* hidden = (const float*)d_in[3];
    const float* Wr_w = (const float*)d_in[4];  const float* Wr_b = (const float*)d_in[5];
    const float* Ur_w = (const float*)d_in[6];  const float* Ur_b = (const float*)d_in[7];
    const float* Wz_w = (const float*)d_in[8];  const float* Wz_b = (const float*)d_in[9];
    const float* Uz_w = (const float*)d_in[10]; const float* Uz_b = (const float*)d_in[11];
    const float* Wh_w = (const float*)d_in[12]; const float* Wh_b = (const float*)d_in[13];
    const float* Uh_w = (const float*)d_in[14]; const float* Uh_b = (const float*)d_in[15];
    const float* out_w = (const float*)d_in[16]; const float* out_b = (const float*)d_in[17];
    const float* bn_g  = (const float*)d_in[18]; const float* bn_b  = (const float*)d_in[19];

    // Workspace carve-up (256B aligned): stats | bf16 copies | z | h_pre
    char* ws = (char*)d_ws;
    size_t off = 0;
    auto carve = [&](size_t bytes) -> char* {
        char* p = ws + off;
        off += (bytes + 255) & ~(size_t)255;
        return p;
    };
    const size_t nXI = (size_t)B_DIM * IN_DIM;   // 4.19M
    const size_t nBH = (size_t)B_DIM * H_DIM;    // 16.78M
    const size_t nWI = (size_t)H_DIM * IN_DIM;   // 4.19M
    const size_t nUH = (size_t)H_DIM * H_DIM;    // 16.78M

    float* stats = (float*)carve(2 * sizeof(float));
    u16* xbf  = (u16*)carve(nXI * 2);
    u16* hbf  = (u16*)carve(nBH * 2);
    u16* wrbf = (u16*)carve(nWI * 2);
    u16* wzbf = (u16*)carve(nWI * 2);
    u16* whbf = (u16*)carve(nWI * 2);
    u16* urbf = (u16*)carve(nUH * 2);
    u16* uzbf = (u16*)carve(nUH * 2);
    u16* uhbf = (u16*)carve(nUH * 2);
    float* zbuf    = (float*)carve(nBH * sizeof(float));
    float* hprebuf = (float*)carve(nBH * sizeof(float));

    hipMemsetAsync(stats, 0, 2 * sizeof(float), stream);

    // Casts to bf16
    cast_f32_to_bf16<<<2048, 256, 0, stream>>>(x,    xbf,  nXI);
    cast_f32_to_bf16<<<4096, 256, 0, stream>>>(hidden, hbf, nBH);
    cast_f32_to_bf16<<<2048, 256, 0, stream>>>(Wr_w, wrbf, nWI);
    cast_f32_to_bf16<<<2048, 256, 0, stream>>>(Wz_w, wzbf, nWI);
    cast_f32_to_bf16<<<2048, 256, 0, stream>>>(Wh_w, whbf, nWI);
    cast_f32_to_bf16<<<4096, 256, 0, stream>>>(Ur_w, urbf, nUH);
    cast_f32_to_bf16<<<4096, 256, 0, stream>>>(Uz_w, uzbf, nUH);
    cast_f32_to_bf16<<<4096, 256, 0, stream>>>(Uh_w, uhbf, nUH);

    // Fused 6-GEMM gate kernel: grid (H/64, B/128)
    dim3 ggrid(H_DIM / 64, B_DIM / 128);
    gru_gates_wmma<<<ggrid, 256, 0, stream>>>(
        xbf, hbf, wrbf, wzbf, whbf, urbf, uzbf, uhbf,
        Wr_b, Ur_b, Wz_b, Uz_b, Wh_b, Uh_b, zbuf, hprebuf);

    // BatchNorm stats
    bn_stats<<<2048, 256, 0, stream>>>(hprebuf, stats);

    // Finalize + projection. d_out = [state (B*H) | o0 (B) | o1 | o2 | o3]
    float* state_out = (float*)d_out;
    float* proj_out  = state_out + nBH;
    gru_finalize<<<B_DIM, 256, 0, stream>>>(zbuf, hprebuf, hidden, stats,
                                            out_w, out_b, bn_g, bn_b,
                                            state_out, proj_out);
}
